// PowerIterationConv_45122926411973
// MI455X (gfx1250) — matrix operand order repaired and checked by reading, not compiled, not run on hardware
//
#include <hip/hip_runtime.h>
#include <hip/hip_bf16.h>

// ---------------------------------------------------------------------------
// Problem: 3 power iterations of a 1024->1024 5x5 circular conv + rescale.
// GEMM view: D[1024 x 121] = W[1024 x 25600] * P[25600 x 121], N padded to 128.
// bf16 WMMA (K=32/instr) with f32 accumulation keeps the GEMM near the
// 23.3 TB/s memory floor; fp32 WMMA (K=4/instr) would be ~8x more matrix ops.
// ---------------------------------------------------------------------------
#define C_CH   1024
#define K5     5
#define TAPS   25            // 5*5
#define H11    11
#define S121   121           // 11*11 spatial positions
#define SPAD   128           // N padded to 8 WMMA tiles
#define KK     25600         // 1024*25 reduction length
#define KSPLIT 4             // K split for parallelism (deterministic reduce)
#define KPART  (KK / KSPLIT) // 6400
#define KTILE  64            // bf16 elements per TDM tile along K (2 WMMA steps)

typedef __attribute__((ext_vector_type(16))) __bf16       v16bf;
typedef __attribute__((ext_vector_type(8)))  float        v8f;
typedef __attribute__((ext_vector_type(4)))  float        v4f;
typedef __attribute__((ext_vector_type(4)))  unsigned int v4u;
typedef __attribute__((ext_vector_type(8)))  int          v8i;
typedef __attribute__((ext_vector_type(4)))  int          v4i;

#if __has_builtin(__builtin_amdgcn_tensor_load_to_lds)
#define HAVE_TDM 1
#else
#define HAVE_TDM 0
#endif

// ---------------------------------------------------------------------------
// A-fragment: 16x32 bf16, contiguous k order (same as B):
//   lanes 0-15 : M=lane,    K = kbase+0..15
//   lanes 16-31: M=lane-16, K = kbase+16..31
// Loaded from f32 W and converted in-register (cvt co-executes with WMMA).
// ---------------------------------------------------------------------------
__device__ inline v16bf load_A_frag(const float* __restrict__ Wf, int co, int kbase, int hi) {
    const float* p = Wf + (size_t)co * KK + kbase + (hi ? 16 : 0);
    v4f f0 = *(const v4f*)(p);          // K + 0..3
    v4f f1 = *(const v4f*)(p + 4);      // K + 4..7
    v4f f2 = *(const v4f*)(p + 8);      // K + 8..11
    v4f f3 = *(const v4f*)(p + 12);     // K + 12..15
    union { v16bf v; __bf16 e[16]; } a;
#pragma unroll
    for (int i = 0; i < 4; ++i) {
        a.e[i]      = (__bf16)f0[i];
        a.e[4 + i]  = (__bf16)f1[i];
        a.e[8 + i]  = (__bf16)f2[i];
        a.e[12 + i] = (__bf16)f3[i];
    }
    return a.v;
}

#if HAVE_TDM
// ---------------------------------------------------------------------------
// TDM: async-load a 2D tile of Pt (all 128 rows x KTILE bf16, row stride KK)
// into LDS at byte offset lds_off. D# per CDNA5 ISA ch.8:
//  group0: [1:0]=count=1 | [63:32]=lds_addr | [120:64]=global tile addr |
//          [127:126]=type=2
//  group1: [17:16]=data_size=1(2B) | dim0=KK | dim1=SPAD | tile0=KTILE |
//          tile1=SPAD | dim0_stride=KK
//  groups 2/3: zero (2-D tensor)
// ---------------------------------------------------------------------------
__device__ inline void tdm_load_B(const __bf16* gsrc, unsigned lds_off) {
    unsigned long long ga = (unsigned long long)(uintptr_t)gsrc;
    v4u g0;
    g0[0] = 1u;                                    // count=1 (valid user D#)
    g0[1] = lds_off;                               // LDS byte address
    g0[2] = (unsigned)(ga & 0xFFFFFFFFu);          // global_addr[31:0]
    g0[3] = (unsigned)((ga >> 32) & 0x01FFFFFFu)   // global_addr[56:32]
          | (2u << 30);                            // type=2 ("image")
    v8i g1;
    g1[0] = 1 << 16;                               // data_size=1 -> 2 bytes
    g1[1] = (KK & 0xFFFF) << 16;                   // tensor_dim0[15:0]
    g1[2] = (KK >> 16) | (SPAD << 16);             // td0[31:16] | td1[15:0]
    g1[3] = (KTILE << 16);                         // td1[31:16]=0 | tile_dim0
    g1[4] = SPAD;                                  // tile_dim1 | tile_dim2=0
    g1[5] = KK;                                    // tensor_dim0_stride[31:0]
    g1[6] = 0;                                     // stride0[47:32] | stride1 lo
    g1[7] = 0;
    v4i gz = {0, 0, 0, 0};
#if defined(__clang_major__) && __clang_major__ >= 23
    v8i gz8 = {0, 0, 0, 0, 0, 0, 0, 0};
    __builtin_amdgcn_tensor_load_to_lds(g0, g1, gz, gz, gz8, 0);
#else
    __builtin_amdgcn_tensor_load_to_lds(g0, g1, gz, gz, 0);
#endif
}
#endif // HAVE_TDM

// ---------------------------------------------------------------------------
// B-fragment direct-from-global fallback: 32x16 bf16, contiguous k order:
//   lanes 0-15 : N=lane, K=kbase+0..15 ; lanes 16-31: N=lane-16, K=kbase+16..31
// ---------------------------------------------------------------------------
__device__ inline v16bf load_B_frag_global(const __bf16* __restrict__ Pt,
                                           int scol, int kbase, int hi) {
    const __bf16* p = Pt + (size_t)scol * KK + kbase + (hi ? 16 : 0);
    return *(const v16bf*)p;
}

// ---------------------------------------------------------------------------
// GEMM: blockIdx.x = m-tile (0..63), blockIdx.y = K-part (0..3).
// 8 waves/block; wave w owns 16x16 tile (mtile, ntile=w) over its K range.
// B tiles stream global->LDS via TDM (double buffered); A streams from
// global f32 with in-register bf16 conversion. Partials per K-part are
// reduced later in a fixed order (determinism: no float atomics).
// ---------------------------------------------------------------------------
__global__ __launch_bounds__(256) void gemm_wmma_bf16(const float* __restrict__ Wf,
                                                      const __bf16* __restrict__ Pt,
                                                      float* __restrict__ U2p) {
    const int lane  = threadIdx.x & 31;
    const int wave  = threadIdx.x >> 5;      // 0..7  -> ntile
    const int mtile = blockIdx.x;            // 0..63
    const int kpart = blockIdx.y;            // 0..3
    const int hi    = lane >> 4;
    const int l16   = lane & 15;
    const int co    = mtile * 16 + l16;      // A row for this lane
    const int scol  = wave * 16 + l16;       // B column for this lane
    const int k0    = kpart * KPART;

    v8f acc = {};

#if HAVE_TDM
    __shared__ __align__(128) __bf16 Blds[2][SPAD * KTILE];  // 2 x 16 KB
    const unsigned lds_off0 = (unsigned)(uintptr_t)(&Blds[0][0]);
    const unsigned lds_off1 = (unsigned)(uintptr_t)(&Blds[1][0]);

    // Prologue: wave 0 DMAs tile 0, waits TENSORcnt, then everyone syncs.
    if (wave == 0) {
        tdm_load_B(Pt + k0, lds_off0);
        __builtin_amdgcn_s_wait_tensorcnt(0);
    }
    __syncthreads();

    for (int t = 0; t < KPART / KTILE; ++t) {          // 100 tiles
        const int kb = k0 + t * KTILE;
        if (wave == 0 && (t + 1) < KPART / KTILE)      // async-prefetch next
            tdm_load_B(Pt + (size_t)(kb + KTILE), (t & 1) ? lds_off0 : lds_off1);

        const __bf16* Bbuf = &Blds[t & 1][0];
        __builtin_prefetch(Wf + (size_t)co * KK + kb + 256, 0, 1);

        v16bf a0 = load_A_frag(Wf, co, kb, hi);
        v16bf b0 = *(const v16bf*)(Bbuf + scol * KTILE + hi * 16);
        acc = __builtin_amdgcn_wmma_f32_16x16x32_bf16(false, a0, false, b0,
                                                      (short)0, acc, false, false);
        v16bf a1 = load_A_frag(Wf, co, kb + 32, hi);
        v16bf b1 = *(const v16bf*)(Bbuf + scol * KTILE + 32 + hi * 16);
        acc = __builtin_amdgcn_wmma_f32_16x16x32_bf16(false, a1, false, b1,
                                                      (short)0, acc, false, false);

        if (wave == 0) __builtin_amdgcn_s_wait_tensorcnt(0);
        __syncthreads();   // next buffer visible; this buffer free to overwrite
    }
#else
    // Fallback: direct global B fragments.
#pragma unroll 2
    for (int kc = 0; kc < KPART; kc += 32) {
        const int k = k0 + kc;
        __builtin_prefetch(Wf + (size_t)co * KK + k + 256, 0, 1);
        v16bf a = load_A_frag(Wf, co, k, hi);
        v16bf b = load_B_frag_global(Pt, scol, k, hi);
        acc = __builtin_amdgcn_wmma_f32_16x16x32_bf16(false, a, false, b,
                                                      (short)0, acc, false, false);
    }
#endif

    // C/D layout: VGPR r, lanes 0-15 -> (M=r, N=lane); lanes 16-31 -> (M=r+8).
    float* dst = U2p + (size_t)kpart * (C_CH * SPAD)
                     + (size_t)(mtile * 16 + hi * 8) * SPAD + wave * 16 + l16;
#pragma unroll
    for (int r = 0; r < 8; ++r) dst[(size_t)r * SPAD] = acc[r];
}

// ---------------------------------------------------------------------------
// Build im2col matrix Pt[s][k] (bf16) from current u [1024][128] (f32).
// k = ci*25 + ky*5 + kx ; circular pad folded in: y = (sy+ky+9) % 11.
// Columns s in [121,128) are zero so padded GEMM columns are exact zeros.
// ---------------------------------------------------------------------------
__global__ __launch_bounds__(256) void build_P(const float* __restrict__ Ucur,
                                               __bf16* __restrict__ Pt) {
    const int idx = blockIdx.x * 256 + threadIdx.x;   // SPAD*KK total
    if (idx >= SPAD * KK) return;
    const int s = idx / KK;
    const int k = idx - s * KK;
    float v = 0.0f;
    if (s < S121) {
        const int ci = k / TAPS;
        const int t  = k - ci * TAPS;
        const int ky = t / K5, kx = t - ky * K5;
        const int sy = s / H11, sx = s - sy * H11;
        const int y = (sy + ky + H11 - 2) % H11;
        const int x = (sx + kx + H11 - 2) % H11;
        v = Ucur[(size_t)ci * SPAD + y * H11 + x];
    }
    Pt[(size_t)s * KK + k] = (__bf16)v;
}

// ---------------------------------------------------------------------------
// Per-channel spatial L2 normalize (and optional sigma), summing `parts`
// partial buffers in fixed order (deterministic). src: [parts][C][row_stride].
// ---------------------------------------------------------------------------
__global__ __launch_bounds__(128) void reduce_normalize(const float* __restrict__ src,
                                                        int row_stride, int parts,
                                                        size_t part_stride,
                                                        float* __restrict__ Ucur,
                                                        float* __restrict__ sigma,
                                                        int write_sigma) {
    const int c = blockIdx.x;
    const int t = threadIdx.x;
    float v = 0.0f;
    if (t < S121) {
        for (int j = 0; j < parts; ++j)
            v += src[(size_t)j * part_stride + (size_t)c * row_stride + t];
    }
    float ss = v * v;
#pragma unroll
    for (int off = 16; off > 0; off >>= 1) ss += __shfl_xor(ss, off, 32);
    __shared__ float red[5];
    if ((t & 31) == 0) red[t >> 5] = ss;
    __syncthreads();
    if (t == 0) {
        float n = sqrtf(red[0] + red[1] + red[2] + red[3]);
        if (write_sigma) sigma[c] = n;
        red[4] = n;
    }
    __syncthreads();
    const float n = red[4];
    Ucur[(size_t)c * SPAD + t] = (t < S121) ? (v / (n + 1e-12f)) : 0.0f;
}

// ---------------------------------------------------------------------------
// out[co,ci,ky,kx] = kernel[co,ci,ky,kx] / sigma[co]   (float4 streaming)
// ---------------------------------------------------------------------------
__global__ __launch_bounds__(256) void scale_kernel(const float* __restrict__ Wf,
                                                    const float* __restrict__ sigma,
                                                    float* __restrict__ out) {
    const size_t i = ((size_t)blockIdx.x * 256 + threadIdx.x) * 4;
    if (i >= (size_t)C_CH * KK) return;
    const int co = (int)(i / KK);          // 25600 % 4 == 0 -> same co for all 4
    const float s = sigma[co];
    v4f w = *(const v4f*)(Wf + i);
    v4f r;
#pragma unroll
    for (int j = 0; j < 4; ++j) r[j] = w[j] / s;
    *(v4f*)(out + i) = r;
}

// ---------------------------------------------------------------------------
// Launch: normalize(u0) ; 3x { build_P ; gemm ; normalize(+sigma) } ; scale.
// Workspace (256B-aligned offsets):
//   Pt   : 128*25600*2 = 6,553,600 B
//   Ucur : 1024*128*4  =   524,288 B
//   U2p  : 4*1024*128*4= 2,097,152 B
//   sigma: 1024*4      =     4,096 B     total ~9.2 MB
// ---------------------------------------------------------------------------
extern "C" void kernel_launch(void* const* d_in, const int* in_sizes, int n_in,
                              void* d_out, int out_size, void* d_ws, size_t ws_size,
                              hipStream_t stream) {
    const float* Wf = (const float*)d_in[0];   // kernel (1024,1024,5,5) f32
    const float* u0 = (const float*)d_in[1];   // u0 (1,1024,11,11) f32
    float* out = (float*)d_out;

    char* ws = (char*)d_ws;
    __bf16* Pt   = (__bf16*)(ws);
    float*  Ucur = (float*)(ws + 6553600);
    float*  U2p  = (float*)(ws + 6553600 + 524288);
    float*  sig  = (float*)(ws + 6553600 + 524288 + 2097152);

    // u = l2_normalize(u0)  (u0 densely packed, row_stride 121)
    reduce_normalize<<<C_CH, 128, 0, stream>>>(u0, S121, 1, 0, Ucur, sig, 0);

    for (int it = 0; it < 3; ++it) {
        build_P<<<(SPAD * KK) / 256, 256, 0, stream>>>(Ucur, Pt);
        gemm_wmma_bf16<<<dim3(C_CH / 16, KSPLIT), 256, 0, stream>>>(Wf, Pt, U2p);
        reduce_normalize<<<C_CH, 128, 0, stream>>>(U2p, SPAD, KSPLIT,
                                                   (size_t)C_CH * SPAD,
                                                   Ucur, sig, it == 2 ? 1 : 0);
    }

    scale_kernel<<<((size_t)C_CH * KK) / 4 / 256, 256, 0, stream>>>(Wf, sig, out);
}
